// MSDeformAttnTransformerEncoderOnly_21689584844814
// MI455X (gfx1250) — matrix operand order repaired and checked by reading, not compiled, not run on hardware
//
#include <hip/hip_runtime.h>
#include <hip/hip_bf16.h>
#include <math.h>

// ---------------------------------------------------------------------------
// MSDeformAttn encoder for MI455X (gfx1250, wave32).
// GEMMs via v_wmma_f32_16x16x32_bf16 with a software-pipelined operand fetch:
// all fragment loads for K-step kt+1 are issued before the 8 WMMAs of kt,
// so matrix math overlaps L2 latency instead of stalling on s_wait_loadcnt.
//  - A (activations, bf16 row-major): 2x global_load_b128 per fragment
//  - B (weights): pre-packed once into fragment-major layout -> contiguous 32B
// ---------------------------------------------------------------------------

typedef __bf16 bf16_t;
typedef __attribute__((ext_vector_type(16))) __bf16 v16bf;
typedef __attribute__((ext_vector_type(8)))  __bf16 v8bf;
typedef __attribute__((ext_vector_type(8)))  float  v8f;

#define N_HEADS 8
#define D_MODEL 256
#define DEPTH   6
#define BATCH   2
#define L_TOK   5440              // 64*64 + 32*32 + 16*16 + 8*8
#define M_ROWS  (BATCH * L_TOK)   // 10880 = 128 * 85

#define BLK_M 128                 // 4 waves along M, 32 rows each
#define BLK_N 128                 // 2 waves along N, 64 cols each

#define SHUF16(lo, hi) __builtin_shufflevector(lo, hi, 0,1,2,3,4,5,6,7,8,9,10,11,12,13,14,15)

// ---------------------------------------------------------------------------
// 1. Pack (B,d,H,W) levels into x[B,L,d] and pos_flat[B,L,d] (+level_embed)
// ---------------------------------------------------------------------------
__global__ void pack_inputs(const float* __restrict__ s0, const float* __restrict__ s1,
                            const float* __restrict__ s2, const float* __restrict__ s3,
                            const float* __restrict__ p0, const float* __restrict__ p1,
                            const float* __restrict__ p2, const float* __restrict__ p3,
                            const float* __restrict__ lvl_emb,
                            float* __restrict__ x, float* __restrict__ pos)
{
    int gid = blockIdx.x * blockDim.x + threadIdx.x;
    if (gid >= M_ROWS * D_MODEL) return;
    int c   = gid & (D_MODEL - 1);
    int row = gid >> 8;                  // b*L + l
    int b   = row / L_TOK;
    int l   = row - b * L_TOK;

    int lvl, H, W, start;
    if      (l < 4096) { lvl = 0; H = 64; W = 64; start = 0;    }
    else if (l < 5120) { lvl = 1; H = 32; W = 32; start = 4096; }
    else if (l < 5376) { lvl = 2; H = 16; W = 16; start = 5120; }
    else               { lvl = 3; H = 8;  W = 8;  start = 5376; }
    int local = l - start;

    const float* s; const float* p;
    if      (lvl == 0) { s = s0; p = p0; }
    else if (lvl == 1) { s = s1; p = p1; }
    else if (lvl == 2) { s = s2; p = p2; }
    else               { s = s3; p = p3; }

    size_t idx = (((size_t)b * D_MODEL + c) * H + (local / W)) * W + (local % W);
    x[gid]   = s[idx];
    pos[gid] = p[idx] + lvl_emb[lvl * D_MODEL + c];
}

// ---------------------------------------------------------------------------
// 2. One-time weight repack f32[K,N] -> bf16 fragment-major:
//    out[(((d*(N/16)+nt)*(K/32)+kt)*32 + lane)*16 + j]
//      = W[d, kt*32 + 8*(lane>>4) + (j&7) + 16*(j>>3), nt*16 + (lane&15)]
//    so each lane's 16-element B fragment is one contiguous 32-byte run.
// ---------------------------------------------------------------------------
__global__ void pack_weights(const float* __restrict__ W, bf16_t* __restrict__ out,
                             int K, int N, int total)
{
    int pidx = blockIdx.x * blockDim.x + threadIdx.x;
    if (pidx >= total) return;
    int j    = pidx & 15;
    int lane = (pidx >> 4) & 31;
    int rest = pidx >> 9;
    int kt   = rest % (K / 32);
    int r2   = rest / (K / 32);
    int nt   = r2 % (N / 16);
    int d    = r2 / (N / 16);
    int k = kt * 32 + 8 * (lane >> 4) + (j & 7) + 16 * (j >> 3);
    int n = nt * 16 + (lane & 15);
    out[pidx] = (bf16_t)W[((size_t)d * K + k) * N + n];
}

// ---------------------------------------------------------------------------
// 3. LayerNorm kernels (one 256-thread block per token row)
// ---------------------------------------------------------------------------
__global__ void ln1_kernel(const float* __restrict__ x, const float* __restrict__ pos,
                           const float* __restrict__ g, const float* __restrict__ b,
                           bf16_t* __restrict__ qb, bf16_t* __restrict__ xb)
{
    __shared__ float s1[256], s2[256];
    int row = blockIdx.x, c = threadIdx.x;
    float v = x[(size_t)row * 256 + c];
    s1[c] = v; s2[c] = v * v;
    __syncthreads();
    for (int s = 128; s > 0; s >>= 1) {
        if (c < s) { s1[c] += s1[c + s]; s2[c] += s2[c + s]; }
        __syncthreads();
    }
    float m   = s1[0] * (1.f / 256.f);
    float var = s2[0] * (1.f / 256.f) - m * m;
    float inv = rsqrtf(var + 1e-5f);
    float x2  = (v - m) * inv * g[c] + b[c];
    qb[(size_t)row * 256 + c] = (bf16_t)(x2 + pos[(size_t)row * 256 + c]);
    xb[(size_t)row * 256 + c] = (bf16_t)v;
}

__global__ void ln2_kernel(const float* __restrict__ x,
                           const float* __restrict__ g, const float* __restrict__ b,
                           bf16_t* __restrict__ x2b)
{
    __shared__ float s1[256], s2[256];
    int row = blockIdx.x, c = threadIdx.x;
    float v = x[(size_t)row * 256 + c];
    s1[c] = v; s2[c] = v * v;
    __syncthreads();
    for (int s = 128; s > 0; s >>= 1) {
        if (c < s) { s1[c] += s1[c + s]; s2[c] += s2[c + s]; }
        __syncthreads();
    }
    float m   = s1[0] * (1.f / 256.f);
    float var = s2[0] * (1.f / 256.f) - m * m;
    float inv = rsqrtf(var + 1e-5f);
    x2b[(size_t)row * 256 + c] = (bf16_t)((v - m) * inv * g[c] + b[c]);
}

// ---------------------------------------------------------------------------
// 4. WMMA GEMM, LDS-free, double-buffered fragments.
//    256 threads = 8 waves; block tile 128x128; wave tile 32x64
//    (2 M-subtiles x 4 N-subtiles = 8 f32 accumulators, 8 wmma per K-step).
// ---------------------------------------------------------------------------
__global__ __launch_bounds__(256)
void gemm_bf16_wmma(const bf16_t* __restrict__ A, const bf16_t* __restrict__ Wp,
                    const float* __restrict__ bias, const float* __restrict__ residual,
                    float* __restrict__ outF, bf16_t* __restrict__ outB,
                    int M, int N, int K, int do_gelu)
{
    const int tid  = threadIdx.x;
    const int lane = tid & 31;
    const int wave = tid >> 5;            // 0..7
    const int wm   = wave & 3;            // 4 M-strips of 32 rows
    const int wn   = wave >> 2;           // 2 N-strips of 64 cols
    const int bm   = blockIdx.x * BLK_M;
    const int bn   = blockIdx.y * BLK_N;
    const int half = lane >> 4;
    const int mloc = lane & 15;
    const int ktiles = K >> 5;

    // per-lane A fragment pointers (two 16B runs per fragment, ISA 7.12.2)
    const bf16_t* aP0 = A + (size_t)(bm + wm * 32 + mloc) * K + 8 * half;
    const bf16_t* aP1 = aP0 + (size_t)16 * K;
    // packed-B fragment base: nt0 = first n-subtile of this wave
    const int nt0 = (bn + wn * 64) >> 4;
    const bf16_t* wp0 = Wp + ((size_t)nt0 * ktiles * 32 + lane) * 16;

    v8f acc[2][4];
#pragma unroll
    for (int s = 0; s < 2; ++s)
#pragma unroll
        for (int t = 0; t < 4; ++t)
#pragma unroll
            for (int e = 0; e < 8; ++e) acc[s][t][e] = 0.f;

    // ---- pipeline prologue: fragments for kt = 0 ----
    v16bf acur0, acur1, bcur[4];
    {
        v8bf lo0 = *(const v8bf*)(aP0);
        v8bf hi0 = *(const v8bf*)(aP0 + 16);
        v8bf lo1 = *(const v8bf*)(aP1);
        v8bf hi1 = *(const v8bf*)(aP1 + 16);
        acur0 = SHUF16(lo0, hi0);
        acur1 = SHUF16(lo1, hi1);
#pragma unroll
        for (int t = 0; t < 4; ++t)
            bcur[t] = *(const v16bf*)(wp0 + (size_t)t * ktiles * 512);
    }

    // ---- steady state: issue kt+1 loads, then 8 WMMAs on kt ----
#pragma unroll 2
    for (int kt = 0; kt < ktiles - 1; ++kt) {
        const bf16_t* aN0 = aP0 + (size_t)(kt + 1) * 32;
        const bf16_t* aN1 = aP1 + (size_t)(kt + 1) * 32;
        v8bf lo0 = *(const v8bf*)(aN0);
        v8bf hi0 = *(const v8bf*)(aN0 + 16);
        v8bf lo1 = *(const v8bf*)(aN1);
        v8bf hi1 = *(const v8bf*)(aN1 + 16);
        v16bf anext0 = SHUF16(lo0, hi0);
        v16bf anext1 = SHUF16(lo1, hi1);
        v16bf bnext[4];
#pragma unroll
        for (int t = 0; t < 4; ++t)
            bnext[t] = *(const v16bf*)(wp0 + ((size_t)t * ktiles + kt + 1) * 512);

        if (kt + 2 < ktiles) {   // global_prefetch_b8 hint two steps ahead
            __builtin_prefetch(aN0 + 32, 0, 1);
            __builtin_prefetch(wp0 + ((size_t)kt + 2) * 512, 0, 1);
        }

#pragma unroll
        for (int t = 0; t < 4; ++t) {
            acc[0][t] = __builtin_amdgcn_wmma_f32_16x16x32_bf16(
                false, acur0, false, bcur[t], (short)0, acc[0][t], false, false);
            acc[1][t] = __builtin_amdgcn_wmma_f32_16x16x32_bf16(
                false, acur1, false, bcur[t], (short)0, acc[1][t], false, false);
        }
        acur0 = anext0;
        acur1 = anext1;
#pragma unroll
        for (int t = 0; t < 4; ++t) bcur[t] = bnext[t];
    }
    // ---- pipeline epilogue: last K-step ----
#pragma unroll
    for (int t = 0; t < 4; ++t) {
        acc[0][t] = __builtin_amdgcn_wmma_f32_16x16x32_bf16(
            false, acur0, false, bcur[t], (short)0, acc[0][t], false, false);
        acc[1][t] = __builtin_amdgcn_wmma_f32_16x16x32_bf16(
            false, acur1, false, bcur[t], (short)0, acc[1][t], false, false);
    }

    // store: C/D layout -> element (m = r + 8*half, n = lane%16) in VGPR r
#pragma unroll
    for (int t = 0; t < 4; ++t) {
        int col = bn + wn * 64 + t * 16 + mloc;
        float bv = bias ? bias[col] : 0.f;
#pragma unroll
        for (int s = 0; s < 2; ++s) {
#pragma unroll
            for (int r = 0; r < 8; ++r) {
                int row = bm + wm * 32 + s * 16 + r + 8 * half;
                float v = acc[s][t][r] + bv;
                if (do_gelu) v = 0.5f * v * (1.f + erff(v * 0.70710678118654752f));
                if (residual) v += residual[(size_t)row * N + col];
                if (outF) outF[(size_t)row * N + col] = v;
                if (outB) outB[(size_t)row * N + col] = (bf16_t)v;
            }
        }
    }
}

// ---------------------------------------------------------------------------
// 5. Softmax over contiguous groups of 32 logits (one wave32 per group)
// ---------------------------------------------------------------------------
__global__ void softmax32(const float* __restrict__ logits, float* __restrict__ prob)
{
    int wid  = (blockIdx.x * blockDim.x + threadIdx.x) >> 5;
    int lane = threadIdx.x & 31;
    float v = logits[(size_t)wid * 32 + lane];
    float m = v;
#pragma unroll
    for (int o = 16; o > 0; o >>= 1) m = fmaxf(m, __shfl_xor(m, o, 32));
    float e = __expf(v - m);
    float s = e;
#pragma unroll
    for (int o = 16; o > 0; o >>= 1) s += __shfl_xor(s, o, 32);
    prob[(size_t)wid * 32 + lane] = e / s;
}

// ---------------------------------------------------------------------------
// 6. Deformable sampling: one wave32 per (b, token, head); lanes = 32 channels
// ---------------------------------------------------------------------------
__global__ void deform_sample(const float* __restrict__ off, const float* __restrict__ attn,
                              const float* __restrict__ val, bf16_t* __restrict__ samp)
{
    int wid  = (blockIdx.x * blockDim.x + threadIdx.x) >> 5;   // over B*L*HEADS
    int lane = threadIdx.x & 31;
    int h  = wid & 7;
    int bl = wid >> 3;                 // b*L + l
    int b  = bl / L_TOK;
    int l  = bl - b * L_TOK;

    int H0, W0, st0;
    if      (l < 4096) { H0 = 64; W0 = 64; st0 = 0;    }
    else if (l < 5120) { H0 = 32; W0 = 32; st0 = 4096; }
    else if (l < 5376) { H0 = 16; W0 = 16; st0 = 5120; }
    else               { H0 = 8;  W0 = 8;  st0 = 5376; }
    int local = l - st0;
    float rx = ((local % W0) + 0.5f) / (float)W0;
    float ry = ((local / W0) + 0.5f) / (float)H0;

    const float* offp = off  + (size_t)bl * 512 + h * 64;      // [lv*16 + p*2 + {x,y}]
    const float* atp  = attn + (size_t)bl * 256 + h * 32;      // [lv*8 + p]

    const int Hs[4] = {64, 32, 16, 8};
    const int Ws[4] = {64, 32, 16, 8};
    const int Ss[4] = {0, 4096, 5120, 5376};

    float acc = 0.f;
#pragma unroll
    for (int lv = 0; lv < 4; ++lv) {
        const int   Hl = Hs[lv], Wl = Ws[lv];
        const float fH = (float)Hl, fW = (float)Wl;
        const float* vbase = val + ((size_t)(b * L_TOK + Ss[lv])) * 256 + h * 32 + lane;
#pragma unroll
        for (int p = 0; p < 8; ++p) {
            float ox = offp[lv * 16 + p * 2 + 0];
            float oy = offp[lv * 16 + p * 2 + 1];
            float aw = atp[lv * 8 + p];
            float sx = (rx + ox / fW) * fW - 0.5f;
            float sy = (ry + oy / fH) * fH - 0.5f;
            float x0 = floorf(sx), y0 = floorf(sy);
#pragma unroll
            for (int dy = 0; dy < 2; ++dy)
#pragma unroll
                for (int dx = 0; dx < 2; ++dx) {
                    float xi = x0 + (float)dx, yi = y0 + (float)dy;
                    float w  = (1.f - fabsf(sx - xi)) * (1.f - fabsf(sy - yi));
                    bool ok  = (xi >= 0.f) && (xi < fW) && (yi >= 0.f) && (yi < fH);
                    int ix = (int)fminf(fmaxf(xi, 0.f), fW - 1.f);
                    int iy = (int)fminf(fmaxf(yi, 0.f), fH - 1.f);
                    float g = vbase[(size_t)(iy * Wl + ix) * 256];
                    acc += aw * (ok ? w : 0.f) * g;
                }
        }
    }
    samp[(size_t)bl * 256 + h * 32 + lane] = (bf16_t)acc;
}

// ---------------------------------------------------------------------------
// Host orchestration
// ---------------------------------------------------------------------------
extern "C" void kernel_launch(void* const* d_in, const int* in_sizes, int n_in,
                              void* d_out, int out_size, void* d_ws, size_t ws_size,
                              hipStream_t stream)
{
    (void)in_sizes; (void)n_in; (void)out_size; (void)ws_size;

    // setup_inputs() dict order (src/pos interleaved):
    const float* src[4] = {(const float*)d_in[0], (const float*)d_in[2],
                           (const float*)d_in[4], (const float*)d_in[6]};
    const float* pin[4] = {(const float*)d_in[1], (const float*)d_in[3],
                           (const float*)d_in[5], (const float*)d_in[7]};
    const float* lvl_emb = (const float*)d_in[8];
    const float* W_off  = (const float*)d_in[9];   const float* b_off  = (const float*)d_in[10];
    const float* W_attn = (const float*)d_in[11];  const float* b_attn = (const float*)d_in[12];
    const float* W_val  = (const float*)d_in[13];  const float* b_val  = (const float*)d_in[14];
    const float* W_out  = (const float*)d_in[15];  const float* b_out  = (const float*)d_in[16];
    const float* ln1_g  = (const float*)d_in[17];  const float* ln1_b  = (const float*)d_in[18];
    const float* W1     = (const float*)d_in[19];  const float* b1     = (const float*)d_in[20];
    const float* W2     = (const float*)d_in[21];  const float* b2     = (const float*)d_in[22];
    const float* ln2_g  = (const float*)d_in[23];  const float* ln2_b  = (const float*)d_in[24];

    // -------- workspace carve-out --------
    char* ws = (char*)d_ws;
    auto take = [&](size_t bytes) -> void* {
        void* p = (void*)ws;
        ws += (bytes + 255) & ~(size_t)255;
        return p;
    };
    bf16_t* wb_off  = (bf16_t*)take((size_t)DEPTH * 256 * 512  * sizeof(bf16_t));
    bf16_t* wb_attn = (bf16_t*)take((size_t)DEPTH * 256 * 256  * sizeof(bf16_t));
    bf16_t* wb_val  = (bf16_t*)take((size_t)DEPTH * 256 * 256  * sizeof(bf16_t));
    bf16_t* wb_out  = (bf16_t*)take((size_t)DEPTH * 256 * 256  * sizeof(bf16_t));
    bf16_t* wb_1    = (bf16_t*)take((size_t)DEPTH * 256 * 1024 * sizeof(bf16_t));
    bf16_t* wb_2    = (bf16_t*)take((size_t)DEPTH * 1024 * 256 * sizeof(bf16_t));
    float*  xbuf    = (float*) take((size_t)M_ROWS * 256 * sizeof(float));
    float*  posbuf  = (float*) take((size_t)M_ROWS * 256 * sizeof(float));
    bf16_t* qb      = (bf16_t*)take((size_t)M_ROWS * 256 * sizeof(bf16_t));
    bf16_t* xb      = (bf16_t*)take((size_t)M_ROWS * 256 * sizeof(bf16_t));
    float*  offbuf  = (float*) take((size_t)M_ROWS * 512 * sizeof(float));
    float*  attnbuf = (float*) take((size_t)M_ROWS * 256 * sizeof(float));
    float*  valbuf  = (float*) take((size_t)M_ROWS * 256 * sizeof(float));
    bf16_t* sampbuf = (bf16_t*)take((size_t)M_ROWS * 256 * sizeof(bf16_t));
    // aliases (temporally disjoint within a layer):
    bf16_t* x2b = qb;                    // q dead after W_off/W_attn GEMMs
    bf16_t* h1b = (bf16_t*)offbuf;       // off dead after sampling; same byte size

    const int nelem = M_ROWS * 256;

    // -------- stage 0: pack inputs + fragment-major weight repack --------
    pack_inputs<<<(nelem + 255) / 256, 256, 0, stream>>>(
        src[0], src[1], src[2], src[3], pin[0], pin[1], pin[2], pin[3],
        lvl_emb, xbuf, posbuf);
    pack_weights<<<(DEPTH*256*512  + 255)/256, 256, 0, stream>>>(W_off,  wb_off,  256, 512,  DEPTH*256*512);
    pack_weights<<<(DEPTH*256*256  + 255)/256, 256, 0, stream>>>(W_attn, wb_attn, 256, 256,  DEPTH*256*256);
    pack_weights<<<(DEPTH*256*256  + 255)/256, 256, 0, stream>>>(W_val,  wb_val,  256, 256,  DEPTH*256*256);
    pack_weights<<<(DEPTH*256*256  + 255)/256, 256, 0, stream>>>(W_out,  wb_out,  256, 256,  DEPTH*256*256);
    pack_weights<<<(DEPTH*256*1024 + 255)/256, 256, 0, stream>>>(W1,     wb_1,    256, 1024, DEPTH*256*1024);
    pack_weights<<<(DEPTH*1024*256 + 255)/256, 256, 0, stream>>>(W2,     wb_2,    1024, 256, DEPTH*1024*256);

    // -------- encoder layers --------
    for (int i = 0; i < DEPTH; ++i) {
        const bf16_t* Wo  = wb_off  + (size_t)i * 256 * 512;
        const bf16_t* Wa  = wb_attn + (size_t)i * 256 * 256;
        const bf16_t* Wv  = wb_val  + (size_t)i * 256 * 256;
        const bf16_t* Wpj = wb_out  + (size_t)i * 256 * 256;
        const bf16_t* Wf1 = wb_1    + (size_t)i * 256 * 1024;
        const bf16_t* Wf2 = wb_2    + (size_t)i * 1024 * 256;

        ln1_kernel<<<M_ROWS, 256, 0, stream>>>(xbuf, posbuf,
                                               ln1_g + i * 256, ln1_b + i * 256, qb, xb);
        // off = q @ W_off + b_off
        gemm_bf16_wmma<<<dim3(M_ROWS/BLK_M, 512/BLK_N), 256, 0, stream>>>(
            qb, Wo, b_off + (size_t)i * 512, nullptr, offbuf, nullptr, M_ROWS, 512, 256, 0);
        // attn logits -> softmax over 32 (in-place)
        gemm_bf16_wmma<<<dim3(M_ROWS/BLK_M, 256/BLK_N), 256, 0, stream>>>(
            qb, Wa, b_attn + (size_t)i * 256, nullptr, attnbuf, nullptr, M_ROWS, 256, 256, 0);
        softmax32<<<(M_ROWS * 256) / 256, 256, 0, stream>>>(attnbuf, attnbuf);
        // val = x @ W_val + b_val
        gemm_bf16_wmma<<<dim3(M_ROWS/BLK_M, 256/BLK_N), 256, 0, stream>>>(
            xb, Wv, b_val + (size_t)i * 256, nullptr, valbuf, nullptr, M_ROWS, 256, 256, 0);
        // deformable bilinear sampling -> samp (bf16)
        deform_sample<<<(M_ROWS * N_HEADS) / 8, 256, 0, stream>>>(offbuf, attnbuf, valbuf, sampbuf);
        // x = x + samp @ W_out + b_out
        gemm_bf16_wmma<<<dim3(M_ROWS/BLK_M, 256/BLK_N), 256, 0, stream>>>(
            sampbuf, Wpj, b_out + (size_t)i * 256, xbuf, xbuf, nullptr, M_ROWS, 256, 256, 0);
        ln2_kernel<<<M_ROWS, 256, 0, stream>>>(xbuf, ln2_g + i * 256, ln2_b + i * 256, x2b);
        // h1 = gelu(x2 @ W1 + b1)  (bf16 out)
        gemm_bf16_wmma<<<dim3(M_ROWS/BLK_M, 1024/BLK_N), 256, 0, stream>>>(
            x2b, Wf1, b1 + (size_t)i * 1024, nullptr, nullptr, h1b, M_ROWS, 1024, 256, 1);
        // x = x + h1 @ W2 + b2   (final layer writes straight to d_out)
        float* xdst = (i == DEPTH - 1) ? (float*)d_out : xbuf;
        gemm_bf16_wmma<<<dim3(M_ROWS/BLK_M, 256/BLK_N), 256, 0, stream>>>(
            h1b, Wf2, b2 + (size_t)i * 256, xbuf, xdst, nullptr, M_ROWS, 256, 1024, 0);
    }
}
